// SchNet_31928786878557
// MI455X (gfx1250) — matrix operand order repaired and checked by reading, not compiled, not run on hardware
//
#include <hip/hip_runtime.h>
#include <hip/hip_bf16.h>

#define N_NODES 10000
#define N_EDGES 320000
#define HD      128
#define GG      50
#define NFLT    128
#define NLAY    6
#define FEATW   (GG + 2*HD)   // 306

typedef _Float16 v16h __attribute__((ext_vector_type(16)));
typedef _Float16 h8   __attribute__((ext_vector_type(8)));
typedef float    v8f  __attribute__((ext_vector_type(8)));

// softplus(x) - log(2): branch-free, raw v_exp_f32/v_log_f32 (base-2 hw ops).
// stable form: max(x,0) + ln(1 + exp(-|x|)) - ln2 ; arg of log is in [1,2].
__device__ __forceinline__ float sspf(float x) {
  const float LOG2E = 1.4426950408889634f;
  const float LN2   = 0.6931471805599453f;
  float t = __builtin_amdgcn_exp2f(-fabsf(x) * LOG2E);
  return fmaxf(x, 0.0f) + LN2 * __builtin_amdgcn_logf(1.0f + t) - LN2;
}

// exp(coeff*d^2) with log2e folded in -> single v_exp_f32
__device__ __forceinline__ float gaussv(float w, int k) {
  const float STEP = 10.0f / 49.0f;
  const float C2   = (-0.5f / (STEP * STEP)) * 1.4426950408889634f;
  float d = w - STEP * (float)k;
  return __builtin_amdgcn_exp2f(C2 * d * d);
}

// ---------------------------------------------------------------- utility
__global__ void zero_kernel(float* __restrict__ p, int n) {
  int i = blockIdx.x * blockDim.x + threadIdx.x;
  if (i < n) p[i] = 0.0f;
}

__global__ void copy_kernel(float* __restrict__ dst, const float* __restrict__ src, int n) {
  int i = blockIdx.x * blockDim.x + threadIdx.x;
  if (i < n) dst[i] = src[i];
}

// ------------------------------------------------- edge init (once/launch)
__global__ void edge_init_kernel(const float* __restrict__ pos,
                                 const int* __restrict__ eidx,
                                 float* __restrict__ ew, float* __restrict__ Cc,
                                 float* __restrict__ cnt) {
  int e = blockIdx.x * blockDim.x + threadIdx.x;
  if (e >= N_EDGES) return;
  int r = eidx[e];
  int c = eidx[N_EDGES + e];
  float dx = pos[r * 3 + 0] - pos[c * 3 + 0];
  float dy = pos[r * 3 + 1] - pos[c * 3 + 1];
  float dz = pos[r * 3 + 2] - pos[c * 3 + 2];
  float w  = sqrtf(dx * dx + dy * dy + dz * dz);
  ew[e] = w;
  Cc[e] = 0.5f * (__cosf(w * 3.14159265358979323846f / 10.0f) + 1.0f);
  atomicAdd(&cnt[r], 1.0f);
}

// ------------------------------------------------- coordinate update edges
__global__ void coord_edge_kernel(const float* __restrict__ ew,
                                  const int* __restrict__ eidx,
                                  const float* __restrict__ h,
                                  const float* __restrict__ posw,
                                  const float* __restrict__ cw,   // [306]
                                  const float* __restrict__ cb,   // [1]
                                  float* __restrict__ sums) {
  int e = blockIdx.x * blockDim.x + threadIdx.x;
  if (e >= N_EDGES) return;
  int r = eidx[e];
  int c = eidx[N_EDGES + e];
  float acc = cb[0];
  float w = ew[e];
#pragma unroll 10
  for (int g = 0; g < GG; ++g) acc += gaussv(w, g) * cw[g];
  const float* hr = h + r * HD;
  const float* hc = h + c * HD;
#pragma unroll 8
  for (int k = 0; k < HD; ++k)
    acc += hr[k] * cw[GG + k] + hc[k] * cw[GG + HD + k];
  float dx = posw[r * 3 + 0] - posw[c * 3 + 0];
  float dy = posw[r * 3 + 1] - posw[c * 3 + 1];
  float dz = posw[r * 3 + 2] - posw[c * 3 + 2];
  atomicAdd(&sums[r * 3 + 0], dx * acc);
  atomicAdd(&sums[r * 3 + 1], dy * acc);
  atomicAdd(&sums[r * 3 + 2], dz * acc);
}

__global__ void pos_update_kernel(float* __restrict__ posw,
                                  const float* __restrict__ sums,
                                  const float* __restrict__ cnt) {
  int i = blockIdx.x * blockDim.x + threadIdx.x;
  if (i >= N_NODES) return;
  float ct  = cnt[i];
  float inv = (ct > 0.0f) ? 1.0f / fmaxf(ct, 1.0f) : 0.0f;
  posw[i * 3 + 0] += sums[i * 3 + 0] * inv;
  posw[i * 3 + 1] += sums[i * 3 + 1] * inv;
  posw[i * 3 + 2] += sums[i * 3 + 2] * inv;
}

// -------------------------------------------------------- fused filter MLP
// Per 16-edge tile per wave:
//   T  = ssp(gauss(ew)[16x64pad] @ mlp1^T + b1)   (WMMA, K=64, B1 from LDS)
//   Wf = (T @ mlp2^T + b2) * C                    (WMMA, K=128, T via LDS)
//   agg[col] += xf[row] * Wf                      (atomic scatter, fused)
__global__ __launch_bounds__(256) void filter_scatter_kernel(
    const float* __restrict__ ew, const float* __restrict__ Cc,
    const int* __restrict__ eidx,
    const float* __restrict__ m1w, const float* __restrict__ m1b,
    const float* __restrict__ m2w, const float* __restrict__ m2b,
    const float* __restrict__ xf, float* __restrict__ agg) {
  __shared__ __align__(16) _Float16 t1[8][16][136];    // 34816 B
  __shared__ __align__(16) _Float16 b1lds[NFLT][72];   // 18432 B, [n][k] padded

  int lane  = threadIdx.x & 31;
  int wave  = threadIdx.x >> 5;
  int eBase = blockIdx.x * 128 + wave * 16;
  int half  = lane >> 4;
  int nloc  = lane & 15;
  int mA    = lane & 15;

  // --- stage mlp1^T into LDS, zero-padded K 50 -> 64 (branch-free loads)
  for (int idx = threadIdx.x; idx < NFLT * 64; idx += 256) {
    int n = idx >> 6;
    int k = idx & 63;
    int kc = (k < GG) ? k : (GG - 1);
    float v = m1w[n * GG + kc];               // always in-bounds
    b1lds[n][k] = (_Float16)((k < GG) ? v : 0.0f);
  }

  // --- A1 fragments: Gaussian expansion of one distance per lane-row
  float w = ew[eBase + mA];
  v16h a1[2];
#pragma unroll
  for (int kb = 0; kb < 2; ++kb) {
#pragma unroll
    for (int e = 0; e < 16; ++e) {
      int k = kb * 32 + half * 16 + e;  // contiguous: k = 16*half + e
      a1[kb][e] = (k < GG) ? (_Float16)gaussv(w, k) : (_Float16)0.0f;
    }
  }
  __syncthreads();

  // --- GEMM1 (+bias, ssp) -> LDS intermediate (per-wave slice)
#pragma unroll
  for (int nt = 0; nt < 8; ++nt) {
    int n = nt * 16 + nloc;
    const h8* bn = (const h8*)(&b1lds[n][0]);  // 144B row stride: 16B aligned
    v8f acc = {};
#pragma unroll
    for (int kb = 0; kb < 2; ++kb) {
      int kbase = kb * 32 + half * 16;
      h8 blo = bn[kbase / 8];
      h8 bhi = bn[kbase / 8 + 1];
      v16h b;
#pragma unroll
      for (int e = 0; e < 8; ++e) { b[e] = blo[e]; b[8 + e] = bhi[e]; }
      acc = __builtin_amdgcn_wmma_f32_16x16x32_f16(false, a1[kb], false, b,
                                                   (short)0, acc, false, false);
    }
    float bias = m1b[n];
#pragma unroll
    for (int r = 0; r < 8; ++r) {
      int m = r + 8 * half;
      t1[wave][m][nt * 16 + nloc] = (_Float16)sspf(acc[r] + bias);
    }
  }
  __syncthreads();

  // --- per-lane edge metadata for the 8 D-rows it owns
  int rowv[8], colv[8];
  float cv[8];
#pragma unroll
  for (int r = 0; r < 8; ++r) {
    int e   = eBase + r + 8 * half;
    rowv[r] = eidx[e];
    colv[r] = eidx[N_EDGES + e];
    cv[r]   = Cc[e];
  }

  // --- A2 fragments: hoisted out of the nt loop (depend only on kb)
  const h8* rowp = (const h8*)(&t1[wave][mA][0]);  // 272B row stride: 16B aligned
  v16h a2[4];
#pragma unroll
  for (int kb = 0; kb < 4; ++kb) {
    h8 lo = rowp[(kb * 32 + half * 16) / 8];
    h8 hi = rowp[(kb * 32 + half * 16) / 8 + 1];
#pragma unroll
    for (int e = 0; e < 8; ++e) { a2[kb][e] = lo[e]; a2[kb][8 + e] = hi[e]; }
  }

  // --- GEMM2 (K=128) + fused cutoff * xf[row] scatter into agg[col]
#pragma unroll
  for (int nt = 0; nt < 8; ++nt) {
    int n = nt * 16 + nloc;
    v8f acc = {};
#pragma unroll
    for (int kb = 0; kb < 4; ++kb) {
      v16h b;
      const float4* bp = (const float4*)(m2w + n * NFLT + kb * 32 + half * 16);
#pragma unroll
      for (int q = 0; q < 4; ++q) {
        float4 f = bp[q];
        b[q * 4 + 0] = (_Float16)f.x; b[q * 4 + 1] = (_Float16)f.y;
        b[q * 4 + 2] = (_Float16)f.z; b[q * 4 + 3] = (_Float16)f.w;
      }
      acc = __builtin_amdgcn_wmma_f32_16x16x32_f16(false, a2[kb], false, b,
                                                   (short)0, acc, false, false);
    }
    float bias = m2b[n];
#pragma unroll
    for (int r = 0; r < 8; ++r) {
      float wf = (acc[r] + bias) * cv[r];
      float v  = wf * xf[rowv[r] * NFLT + n];
      atomicAdd(&agg[colv[r] * NFLT + n], v);
    }
  }
}

// --------------------------------------------------- node GEMM (WMMA, 128K)
// Y[rows,128] = op(X[rows,128] @ Wt^T (+bias)) ; Wt is [128 out, 128 in]
template <int BIAS, int SSP, int ACCUM>
__global__ __launch_bounds__(256) void node_gemm_kernel(
    const float* __restrict__ X, const float* __restrict__ Wt,
    const float* __restrict__ bias, float* __restrict__ Y, int nrows) {
  int lane  = threadIdx.x & 31;
  int wave  = threadIdx.x >> 5;
  int rBase = blockIdx.x * 128 + wave * 16;
  int half  = lane >> 4;
  int nloc  = lane & 15;
  int mA    = lane & 15;

  int  gr    = rBase + mA;
  bool rowok = gr < nrows;
  const float* xrow = X + (rowok ? gr : 0) * HD;

  v16h a[4];
#pragma unroll
  for (int kb = 0; kb < 4; ++kb) {
    const float4* p0 = (const float4*)(xrow + kb * 32 + half * 16);
    const float4* p1 = (const float4*)(xrow + kb * 32 + half * 16 + 8);
    float4 f0 = p0[0], f1 = p0[1], f2 = p1[0], f3 = p1[1];
    a[kb][0]  = (_Float16)f0.x; a[kb][1]  = (_Float16)f0.y;
    a[kb][2]  = (_Float16)f0.z; a[kb][3]  = (_Float16)f0.w;
    a[kb][4]  = (_Float16)f1.x; a[kb][5]  = (_Float16)f1.y;
    a[kb][6]  = (_Float16)f1.z; a[kb][7]  = (_Float16)f1.w;
    a[kb][8]  = (_Float16)f2.x; a[kb][9]  = (_Float16)f2.y;
    a[kb][10] = (_Float16)f2.z; a[kb][11] = (_Float16)f2.w;
    a[kb][12] = (_Float16)f3.x; a[kb][13] = (_Float16)f3.y;
    a[kb][14] = (_Float16)f3.z; a[kb][15] = (_Float16)f3.w;
  }

#pragma unroll
  for (int nt = 0; nt < 8; ++nt) {
    int n = nt * 16 + nloc;
    v8f acc = {};
#pragma unroll
    for (int kb = 0; kb < 4; ++kb) {
      v16h b;
      const float4* bp = (const float4*)(Wt + n * HD + kb * 32 + half * 16);
#pragma unroll
      for (int q = 0; q < 4; ++q) {
        float4 f = bp[q];
        b[q * 4 + 0] = (_Float16)f.x; b[q * 4 + 1] = (_Float16)f.y;
        b[q * 4 + 2] = (_Float16)f.z; b[q * 4 + 3] = (_Float16)f.w;
      }
      acc = __builtin_amdgcn_wmma_f32_16x16x32_f16(false, a[kb], false, b,
                                                   (short)0, acc, false, false);
    }
    float bb = BIAS ? bias[n] : 0.0f;
#pragma unroll
    for (int r = 0; r < 8; ++r) {
      int grow = rBase + r + 8 * half;
      if (grow < nrows) {
        float y = acc[r] + bb;
        if (SSP) y = sspf(y);
        if (ACCUM) Y[grow * HD + n] += y;
        else       Y[grow * HD + n] = y;
      }
    }
  }
}

// ----------------------------------------------------------------- driver
extern "C" void kernel_launch(void* const* d_in, const int* in_sizes, int n_in,
                              void* d_out, int out_size, void* d_ws, size_t ws_size,
                              hipStream_t stream) {
  (void)in_sizes; (void)n_in; (void)out_size; (void)ws_size;
  const float* z    = (const float*)d_in[0];
  const float* pos  = (const float*)d_in[1];
  const int*   eidx = (const int*)d_in[2];
  const float* m1w  = (const float*)d_in[3];
  const float* m1b  = (const float*)d_in[4];
  const float* m2w  = (const float*)d_in[5];
  const float* m2b  = (const float*)d_in[6];
  const float* l1w  = (const float*)d_in[7];
  const float* l2w  = (const float*)d_in[8];
  const float* l2b  = (const float*)d_in[9];
  const float* lw   = (const float*)d_in[10];
  const float* lb   = (const float*)d_in[11];
  const float* cwgt = (const float*)d_in[12];
  const float* cbia = (const float*)d_in[13];

  float* ws   = (float*)d_ws;
  float* posw = ws;                       // 30000
  float* hwk  = ws + 30000;               // 1,280,000
  float* ew   = ws + 1310000;             // 320,000
  float* Cc   = ws + 1630000;             // 320,000
  float* cnt  = ws + 1950000;             // 10,000
  float* sums = ws + 1960000;             // 30,000
  float* xf   = ws + 1990000;             // 1,280,000
  float* agg  = ws + 3270000;             // 1,280,000
  float* tmp  = ws + 4550000;             // 1,280,000

  const int T = 256;
  const int gE   = (N_EDGES + T - 1) / T;          // 1250
  const int gN   = (N_NODES + T - 1) / T;          // 40
  const int gNH  = (N_NODES * HD + T - 1) / T;     // 5000
  const int gGmE = N_EDGES / 128;                  // 2500 (exact)
  const int gGmN = (N_NODES + 127) / 128;          // 79

  // init: working copies, distances, cutoff, row counts
  copy_kernel<<<gN * 3, T, 0, stream>>>(posw, pos, N_NODES * 3);
  copy_kernel<<<gNH, T, 0, stream>>>(hwk, z, N_NODES * HD);
  zero_kernel<<<gN, T, 0, stream>>>(cnt, N_NODES);
  edge_init_kernel<<<gE, T, 0, stream>>>(pos, eidx, ew, Cc, cnt);

  for (int l = 0; l < NLAY; ++l) {
    // ---- coordinate update (uses h before this layer's interaction)
    zero_kernel<<<gN * 3, T, 0, stream>>>(sums, N_NODES * 3);
    coord_edge_kernel<<<gE, T, 0, stream>>>(ew, eidx, hwk, posw,
                                            cwgt + l * FEATW, cbia + l, sums);
    pos_update_kernel<<<gN, T, 0, stream>>>(posw, sums, cnt);

    // ---- CFConv interaction
    node_gemm_kernel<0, 0, 0><<<gGmN, T, 0, stream>>>(
        hwk, l1w + l * NFLT * HD, nullptr, xf, N_NODES);
    zero_kernel<<<gNH, T, 0, stream>>>(agg, N_NODES * NFLT);
    filter_scatter_kernel<<<gGmE, T, 0, stream>>>(
        ew, Cc, eidx,
        m1w + l * NFLT * GG, m1b + l * NFLT,
        m2w + l * NFLT * NFLT, m2b + l * NFLT,
        xf, agg);
    node_gemm_kernel<1, 1, 0><<<gGmN, T, 0, stream>>>(
        agg, l2w + l * HD * NFLT, l2b + l * HD, tmp, N_NODES);
    node_gemm_kernel<1, 0, 1><<<gGmN, T, 0, stream>>>(
        tmp, lw + l * HD * HD, lb + l * HD, hwk, N_NODES);
  }

  // outputs: (pos, h) concatenated
  float* out = (float*)d_out;
  copy_kernel<<<gN * 3, T, 0, stream>>>(out, posw, N_NODES * 3);
  copy_kernel<<<gNH, T, 0, stream>>>(out + N_NODES * 3, hwk, N_NODES * HD);
}
